// FlatSquare7x7NNUEv2_47519518163398
// MI455X (gfx1250) — matrix operand order repaired and checked by reading, not compiled, not run on hardware
//
#include <hip/hip_runtime.h>
#include <cstdint>
#include <cstddef>

typedef _Float16 v16h __attribute__((ext_vector_type(16)));
typedef _Float16 v8h  __attribute__((ext_vector_type(8)));
typedef float    v8f  __attribute__((ext_vector_type(8)));

#define AS1 __attribute__((address_space(1)))   // global
#define AS3 __attribute__((address_space(3)))   // LDS

#define NTHREADS 256
#define TM 32
#define QMAX8F (127.0f / 128.0f)

// ---------------- workspace layout (bytes) ----------------
static constexpr size_t WSLOT    = 128u * 1024u;           // 14 x f16 [N][Kpad] slots
static constexpr size_t BIAS_OFF = 14u * WSLOT;            // 14 x 512B f32 bias
static constexpr size_t HEAD_OFF = 2u * 1024u * 1024u;
static constexpr size_t HW0_OFF  = HEAD_OFF + 0;           // 32*128 f32
static constexpr size_t HW1_OFF  = HEAD_OFF + 16384;       // 32*32  f32
static constexpr size_t HW2_OFF  = HEAD_OFF + 20480;       // 3*32   f32
static constexpr size_t HB0_OFF  = HEAD_OFF + 24576;       // 32 f32
static constexpr size_t HB1_OFF  = HEAD_OFF + 24832;       // 32 f32
static constexpr size_t HB2_OFF  = HEAD_OFF + 25088;       // 3  f32
static constexpr size_t FEAT_OFF = 3u * 1024u * 1024u;     // featC [n][128] f32, then featM

struct Ptrs { const void* p[96]; };
struct MapW { const void* W[7]; const void* B[7]; };

// ---------------- weight prep: fold BN, convert to f16 [N][Kpad] ----------------
__global__ __launch_bounds__(NTHREADS) void prep_kernel(Ptrs in, char* ws) {
    const int blk = blockIdx.x;
    const int tid = threadIdx.x;
    if (blk < 14) {
        const int m = blk / 7, l = blk % 7;
        const int wbase = m ? 39 : 1;
        const int bbase = m ? 46 : 8;
        const int gbase = m ? 53 : 15;
        const int tbase = m ? 59 : 21;
        const int ubase = m ? 65 : 27;
        const int vbase = m ? 71 : 33;
        const float* W  = (const float*)in.p[wbase + l];
        const float* Bb = (const float*)in.p[bbase + l];
        const int N  = (l == 6) ? 32 : 128;
        const int K  = (l == 0) ? 8 : ((l == 2) ? (m ? 256 : 512) : ((l == 4) ? 512 : 128));
        const int Kp = (l == 0) ? 32 : K;
        _Float16* Wo = (_Float16*)(ws + (size_t)(m * 7 + l) * WSLOT);
        float*    Bo = (float*)(ws + BIAS_OFF + (size_t)(m * 7 + l) * 512);
        const float *g = nullptr, *bt = nullptr, *mu = nullptr, *vr = nullptr;
        if (l >= 1) {
            g  = (const float*)in.p[gbase + l - 1];
            bt = (const float*)in.p[tbase + l - 1];
            mu = (const float*)in.p[ubase + l - 1];
            vr = (const float*)in.p[vbase + l - 1];
        }
        for (int o = tid; o < N; o += NTHREADS) {
            float s, bias;
            if (l >= 1) { s = g[o] * rsqrtf(vr[o] + 1e-5f); bias = (Bb[o] - mu[o]) * s + bt[o]; }
            else        { s = 1.0f;                         bias = Bb[o]; }
            Bo[o] = bias;
            for (int k = 0; k < Kp; ++k)
                Wo[(size_t)o * Kp + k] = (_Float16)((k < K) ? W[(size_t)o * K + k] * s : 0.0f);
        }
    } else {
        const float* vw0 = (const float*)in.p[77];
        const float* vw1 = (const float*)in.p[78];
        const float* vw2 = (const float*)in.p[79];
        const float* vb0 = (const float*)in.p[80];
        const float* vb1 = (const float*)in.p[81];
        const float* vb2 = (const float*)in.p[82];
        float* w0 = (float*)(ws + HW0_OFF);
        float* w1 = (float*)(ws + HW1_OFF);
        float* w2 = (float*)(ws + HW2_OFF);
        float* b0 = (float*)(ws + HB0_OFF);
        float* b1 = (float*)(ws + HB1_OFF);
        float* b2 = (float*)(ws + HB2_OFF);
        for (int i = tid; i < 32 * 128; i += NTHREADS) {
            float q = fminf(fmaxf(rintf(vw0[i] * 256.0f), -128.0f), 127.0f);
            w0[i] = q * (1.0f / 256.0f);
        }
        for (int i = tid; i < 32 * 32; i += NTHREADS) {
            float q = fminf(fmaxf(rintf(vw1[i] * 128.0f), -128.0f), 127.0f);
            w1[i] = q * (1.0f / 128.0f);
        }
        for (int i = tid; i < 3 * 32; i += NTHREADS) {
            float q = fminf(fmaxf(rintf(vw2[i] * 128.0f), -128.0f), 127.0f);
            w2[i] = q * (1.0f / 128.0f);
        }
        for (int i = tid; i < 32; i += NTHREADS) {
            b0[i] = rintf(vb0[i] * 32768.0f) * (1.0f / 32768.0f);
            b1[i] = rintf(vb1[i] * 16384.0f) * (1.0f / 16384.0f);
        }
        if (tid < 3) b2[tid] = rintf(vb2[tid] * 16384.0f) * (1.0f / 16384.0f);
    }
}

// ---------------- cheap mish: x * tanh(softplus(x)) = x*(e^2+2e)/(e^2+2e+2) ----------------
__device__ inline float mishf(float x) {
    const float e   = __expf(fminf(x, 20.0f));
    const float num = e * (e + 2.0f);
    return x * (num * __builtin_amdgcn_rcpf(num + 2.0f));
}

// ---------------- WMMA fragment helpers (gfx1250 16x16x32 f16 layouts) ----------------
// A [M][lda] f16 row-major in LDS. Lane (half=lane>>4, r=lane&15) holds row,
// K = kb + half*8 + {0..7} and K = kb + 16 + half*8 + {0..7}.
__device__ inline v16h load_a_frag(const AS3 _Float16* A, int lda, int row, int kb, int half) {
    const AS3 _Float16* p = A + (size_t)row * lda + kb + half * 8;
    v8h lo = *(const AS3 v8h*)p;
    v8h hi = *(const AS3 v8h*)(p + 16);
    v16h r;
#pragma unroll
    for (int j = 0; j < 8; ++j) { r[j] = lo[j]; r[j + 8] = hi[j]; }
    return r;
}

// W [N][K] f16 row-major in global. Lane holds col, K = kb + half*8 + {0..7} and K = kb + 16 + half*8 + {0..7}.
__device__ inline v16h load_b_frag(const AS1 _Float16* W, int ldw, int col, int kb, int half) {
    const AS1 _Float16* p = W + (size_t)col * ldw + kb + half * 8;
    v8h lo = *(const AS1 v8h*)p;
    v8h hi = *(const AS1 v8h*)(p + 16);
    v16h r;
#pragma unroll
    for (int j = 0; j < 8; ++j) { r[j] = lo[j]; r[j + 8] = hi[j]; }
    return r;
}

// One conv layer as GEMM: Out[M][128] = act( A[M][K] x W[128][K]^T + bias ), Ntiles = 8.
__device__ inline void gemm_layer(const AS3 _Float16* A, int lda, int Mtiles,
                                  const AS1 _Float16* W, int K,
                                  const AS1 float* bias, AS3 _Float16* Out, int ldo,
                                  int lane, int wid, bool domish) {
    const int half = lane >> 4, r = lane & 15;
    const int tiles = Mtiles * 8;
    for (int t = wid; t < tiles; t += 8) {
        const int mt = t >> 3, nt = t & 7;
        v8f acc = {};
        // software-pipelined K loop: prefetch next fragments before current WMMA
        v16h a = load_a_frag(A, lda, mt * 16 + r, 0, half);
        v16h b = load_b_frag(W, K, nt * 16 + r, 0, half);
        for (int kb = 32; kb < K; kb += 32) {
            v16h a2 = load_a_frag(A, lda, mt * 16 + r, kb, half);
            v16h b2 = load_b_frag(W, K, nt * 16 + r, kb, half);
            acc = __builtin_amdgcn_wmma_f32_16x16x32_f16(false, a, false, b,
                                                         (short)0, acc, false, false);
            a = a2; b = b2;
        }
        acc = __builtin_amdgcn_wmma_f32_16x16x32_f16(false, a, false, b,
                                                     (short)0, acc, false, false);
        const int nc = nt * 16 + r;
        const float bb = bias[nc];
#pragma unroll
        for (int j = 0; j < 8; ++j) {
            const int mrow = mt * 16 + (half ? 8 : 0) + j;
            float v = acc[j] + bb;
            if (domish) v = mishf(v);
            Out[(size_t)mrow * ldo + nc] = (_Float16)v;
        }
    }
}

// ---------------- fused mapping pipeline (7 layers, LDS-resident) ----------------
// MAP==0: corner 4x4 patches; MAP==1: middle 4x3 patches.
template <int MAP>
__global__ __launch_bounds__(NTHREADS) void mapping_kernel(MapW mw, const float* __restrict__ xin,
                                                           float* __restrict__ feat, int n) {
    constexpr int P0  = (MAP == 0) ? 9 : 6;     // L0 output positions (3x3 / 3x2)
    constexpr int OW0 = (MAP == 0) ? 3 : 2;     // L0 output cols
    constexpr int M0  = TM * P0;
    constexpr int K2  = (MAP == 0) ? 512 : 256; // L2 im2col K
    constexpr int KW2 = (MAP == 0) ? 2 : 1;     // L2 kernel width
    constexpr int XYE = (MAP == 0) ? 65536 : 32768;

    extern __shared__ _Float16 sm[];
    AS3 _Float16* X = (AS3 _Float16*)sm;
    AS3 _Float16* Y = X + XYE;

    const AS1 float* xg = (const AS1 float*)xin;
    AS1 float* fg = (AS1 float*)feat;

    const int tid  = threadIdx.x;
    const int lane = tid & 31;
    const int wid  = __builtin_amdgcn_readfirstlane(tid >> 5);  // wave-uniform -> SGPR
    const int p0   = blockIdx.x * TM;
    // All TM=32 patches of this block belong to one rotation branch (n % 32 == 0).
    const int br = p0 / n;
    const int s0 = p0 - br * n;

    // ---- build L0 im2col A (patch extraction + rot90) into Y: [M0][32], K = cin*4+ky*2+kx
    for (int idx = tid; idx < M0 * 32; idx += NTHREADS) {
        const int mrow = idx >> 5, kk = idx & 31;
        float val = 0.0f;
        if (kk < 8) {
            const int q = mrow / P0, pos = mrow - q * P0;
            const int oy = pos / OW0, ox = pos - oy * OW0;
            const int cin = kk >> 2, ky = (kk >> 1) & 1, kx = kk & 1;
            const int rr = oy + ky, cc = ox + kx;  // rotated-frame coords
            const int s = s0 + q;
            int y0, x0, y, xc;
            if (MAP == 0) {
                switch (br) {
                    case 0:  y0 = 0; x0 = 0; y = rr;     xc = cc;     break;
                    case 1:  y0 = 0; x0 = 3; y = cc;     xc = 3 - rr; break;
                    case 2:  y0 = 3; x0 = 3; y = 3 - rr; xc = 3 - cc; break;
                    default: y0 = 3; x0 = 0; y = 3 - cc; xc = rr;     break;
                }
            } else {
                switch (br) {
                    case 0:  y0 = 0; x0 = 2; y = rr;     xc = cc;     break;
                    case 1:  y0 = 2; x0 = 3; y = cc;     xc = 3 - rr; break;
                    case 2:  y0 = 3; x0 = 2; y = 3 - rr; xc = 2 - cc; break;
                    default: y0 = 2; x0 = 0; y = 2 - cc; xc = rr;     break;
                }
            }
            const int yy = y0 + y, xx = x0 + xc;
            val = xg[(((size_t)s * 2 + cin) * 7 + yy) * 7 + xx];
        }
        Y[(size_t)mrow * 32 + kk] = (_Float16)val;
    }
    __syncthreads();

    const AS1 _Float16* W0 = (const AS1 _Float16*)mw.W[0]; const AS1 float* B0 = (const AS1 float*)mw.B[0];
    const AS1 _Float16* W1 = (const AS1 _Float16*)mw.W[1]; const AS1 float* B1 = (const AS1 float*)mw.B[1];
    const AS1 _Float16* W2 = (const AS1 _Float16*)mw.W[2]; const AS1 float* B2 = (const AS1 float*)mw.B[2];
    const AS1 _Float16* W3 = (const AS1 _Float16*)mw.W[3]; const AS1 float* B3 = (const AS1 float*)mw.B[3];
    const AS1 _Float16* W4 = (const AS1 _Float16*)mw.W[4]; const AS1 float* B4 = (const AS1 float*)mw.B[4];
    const AS1 _Float16* W5 = (const AS1 _Float16*)mw.W[5]; const AS1 float* B5 = (const AS1 float*)mw.B[5];
    const AS1 _Float16* W6 = (const AS1 _Float16*)mw.W[6]; const AS1 float* B6 = (const AS1 float*)mw.B[6];

    // L0: conv2x2 (GEMM, K padded to 32) + mish -> X [M0][128]
    gemm_layer(Y, 32, M0 / 16, W0, 32, B0, X, 128, lane, wid, true);
    __syncthreads();
    // L1: 1x1 + BN + mish -> Y [M0][128]
    gemm_layer(X, 128, M0 / 16, W1, 128, B1, Y, 128, lane, wid, true);
    __syncthreads();

    // im2col for L2: Y[q][3 x OW0 grid][128] -> X[q*4+op][K2], K = cin*(2*KW2)+ky*KW2+kx
    for (int idx = tid; idx < TM * 4 * K2; idx += NTHREADS) {
        const int mrow = idx / K2, k = idx - mrow * K2;
        const int q = mrow >> 2, op = mrow & 3;
        const int py = op >> 1, px = op & 1;
        const int cin = k / (2 * KW2), kr = k - cin * (2 * KW2);
        const int ky = kr / KW2, kx = kr - ky * KW2;
        const int pos = (py + ky) * OW0 + (px + kx);
        X[idx] = Y[(size_t)(q * P0 + pos) * 128 + cin];
    }
    __syncthreads();
    // L2: conv (2x2 / 2x1) + BN + mish -> Y [TM*4][128]
    gemm_layer(X, K2, (TM * 4) / 16, W2, K2, B2, Y, 128, lane, wid, true);
    __syncthreads();
    // L3: 1x1 + BN + mish -> X [TM*4][128]
    gemm_layer(Y, 128, (TM * 4) / 16, W3, 128, B3, X, 128, lane, wid, true);
    __syncthreads();

    // im2col for L4: X[q][2x2 grid][128] -> Y[q][512], K = cin*4 + (ky*2+kx)
    for (int idx = tid; idx < TM * 512; idx += NTHREADS) {
        const int q = idx >> 9, k = idx & 511;
        const int cin = k >> 2, kr = k & 3;
        Y[idx] = X[(size_t)(q * 4 + kr) * 128 + cin];
    }
    __syncthreads();
    // L4: conv2x2 + BN + mish -> X [TM][128]
    gemm_layer(Y, 512, TM / 16, W4, 512, B4, X, 128, lane, wid, true);
    __syncthreads();
    // L5: 1x1 + BN + mish -> Y [TM][128]
    gemm_layer(X, 128, TM / 16, W5, 128, B5, Y, 128, lane, wid, true);
    __syncthreads();

    // L6: 1x1, N=32, BN (no mish), clip + fq, scatter to feature buffer
    {
        const int half = lane >> 4, r = lane & 15;
        const int tiles = (TM / 16) * 2;
        for (int t = wid; t < tiles; t += 8) {
            const int mt = t >> 1, nt = t & 1;
            v8f acc = {};
            v16h a = load_a_frag(Y, 128, mt * 16 + r, 0, half);
            v16h b = load_b_frag(W6, 128, nt * 16 + r, 0, half);
            for (int kb = 32; kb < 128; kb += 32) {
                v16h a2 = load_a_frag(Y, 128, mt * 16 + r, kb, half);
                v16h b2 = load_b_frag(W6, 128, nt * 16 + r, kb, half);
                acc = __builtin_amdgcn_wmma_f32_16x16x32_f16(false, a, false, b,
                                                             (short)0, acc, false, false);
                a = a2; b = b2;
            }
            acc = __builtin_amdgcn_wmma_f32_16x16x32_f16(false, a, false, b,
                                                         (short)0, acc, false, false);
            const int nc = nt * 16 + r;
            const float bb = B6[nc];
#pragma unroll
            for (int j = 0; j < 8; ++j) {
                const int mrow = mt * 16 + (half ? 8 : 0) + j;
                float v = acc[j] + bb;
                v = fminf(fmaxf(v, -1.0f), QMAX8F);
                v = rintf(v * 128.0f) * (1.0f / 128.0f);
                const int s = s0 + mrow;
                fg[(size_t)s * 128 + br * 32 + nc] = v;
            }
        }
    }
}

// ---------------- quantized value head + policy zeros ----------------
__global__ __launch_bounds__(NTHREADS) void head_kernel(const char* __restrict__ ws,
                                                        const float* __restrict__ featC,
                                                        const float* __restrict__ featM,
                                                        float* __restrict__ out, int n) {
    __shared__ float sw0[32 * 128];
    __shared__ float sw1[32 * 32];
    __shared__ float sw2[3 * 32];
    __shared__ float sb0[32], sb1[32], sb2[3];
    const float* w0 = (const float*)(ws + HW0_OFF);
    const float* w1 = (const float*)(ws + HW1_OFF);
    const float* w2 = (const float*)(ws + HW2_OFF);
    const float* b0 = (const float*)(ws + HB0_OFF);
    const float* b1 = (const float*)(ws + HB1_OFF);
    const float* b2 = (const float*)(ws + HB2_OFF);
    for (int i = threadIdx.x; i < 32 * 128; i += NTHREADS) sw0[i] = w0[i];
    for (int i = threadIdx.x; i < 32 * 32; i += NTHREADS)  sw1[i] = w1[i];
    for (int i = threadIdx.x; i < 3 * 32; i += NTHREADS)   sw2[i] = w2[i];
    if (threadIdx.x < 32) { sb0[threadIdx.x] = b0[threadIdx.x]; sb1[threadIdx.x] = b1[threadIdx.x]; }
    if (threadIdx.x < 3)  sb2[threadIdx.x] = b2[threadIdx.x];
    __syncthreads();

    const int s = blockIdx.x * NTHREADS + threadIdx.x;
    if (s < n) {
        float h1[32];
#pragma unroll
        for (int o = 0; o < 32; ++o) h1[o] = sb0[o];
        for (int j = 0; j < 128; ++j) {
            float f = featC[(size_t)s * 128 + j] + featM[(size_t)s * 128 + j];
            f = fminf(fmaxf(f, -1.0f), QMAX8F);
#pragma unroll
            for (int o = 0; o < 32; ++o) h1[o] += f * sw0[o * 128 + j];
        }
        float h2[32];
#pragma unroll
        for (int o = 0; o < 32; ++o) h2[o] = sb1[o];
        for (int j = 0; j < 32; ++j) {
            const float f = fminf(fmaxf(h1[j], 0.0f), QMAX8F);
#pragma unroll
            for (int o = 0; o < 32; ++o) h2[o] += f * sw1[o * 32 + j];
        }
        float h3[3];
#pragma unroll
        for (int o = 0; o < 3; ++o) {
            float a = sb2[o];
            for (int j = 0; j < 32; ++j) {
                const float f = fminf(fmaxf(h2[j], 0.0f), QMAX8F);
                a += f * sw2[o * 32 + j];
            }
            h3[o] = a;
        }
        out[(size_t)s * 3 + 0] = h3[0];
        out[(size_t)s * 3 + 1] = h3[1];
        out[(size_t)s * 3 + 2] = h3[2];
        float* pol = out + (size_t)3 * n + (size_t)s * 49;
        for (int i = 0; i < 49; ++i) pol[i] = 0.0f;
    }
}

// ---------------- launch ----------------
extern "C" void kernel_launch(void* const* d_in, const int* in_sizes, int n_in,
                              void* d_out, int out_size, void* d_ws, size_t ws_size,
                              hipStream_t stream) {
    (void)out_size; (void)ws_size;
    const int n = in_sizes[0] / (2 * 7 * 7);
    char* ws = (char*)d_ws;

    Ptrs P{};
    for (int i = 0; i < n_in && i < 96; ++i) P.p[i] = d_in[i];
    prep_kernel<<<15, NTHREADS, 0, stream>>>(P, ws);

    float* featC = (float*)(ws + FEAT_OFF);
    float* featM = featC + (size_t)n * 128;

    MapW m0{}, m1{};
    for (int l = 0; l < 7; ++l) {
        m0.W[l] = ws + (size_t)l * WSLOT;
        m0.B[l] = ws + BIAS_OFF + (size_t)l * 512;
        m1.W[l] = ws + (size_t)(7 + l) * WSLOT;
        m1.B[l] = ws + BIAS_OFF + (size_t)(7 + l) * 512;
    }

    const float* x = (const float*)d_in[0];
    const int blocks = (4 * n) / TM;  // n is a multiple of TM
    mapping_kernel<0><<<blocks, NTHREADS, 2 * 65536 * 2, stream>>>(m0, x, featC, n);
    mapping_kernel<1><<<blocks, NTHREADS, 2 * 32768 * 2, stream>>>(m1, x, featM, n);

    head_kernel<<<(n + NTHREADS - 1) / NTHREADS, NTHREADS, 0, stream>>>(ws, featC, featM,
                                                                        (float*)d_out, n);
}
